// ExperimentalAttention_63745904607560
// MI455X (gfx1250) — matrix operand order repaired and checked by reading, not compiled, not run on hardware
//
#include <hip/hip_runtime.h>

// ---------------------------------------------------------------------------
// ExperimentalAttention for MI455X (gfx1250, wave32, WMMA bf16 16x16x32)
// B=2 S=2048 D=2048 H=16 HD=128.  ~200 GFLOP vs ~200MB data -> compute bound;
// all GEMM stages go through v_wmma_f32_16x16x32_bf16, fp32 accumulation.
// Working set is L2-resident (192MB) -> direct global B128 fragment loads.
// GEMM wave tile 64x64 (4x4 WMMA) -> 32 flop/byte from L2.
// ---------------------------------------------------------------------------

typedef __attribute__((ext_vector_type(16))) __bf16 v16bf;
typedef __attribute__((ext_vector_type(8)))  float  v8f;
typedef __attribute__((ext_vector_type(4)))  unsigned int u32x4;
typedef unsigned short ushort_t;

static constexpr int Bc = 2, Sc = 2048, Dc = 2048, Hc = 16, HDc = 128;
static constexpr float EPSc = 1e-6f;

union Frag { v16bf v; u32x4 q[2]; };

static __device__ __forceinline__ ushort_t f2bf(float f) {
  union { float f; unsigned int u; } x; x.f = f;
  unsigned int r = x.u + 0x7FFFu + ((x.u >> 16) & 1u);   // round-to-nearest-even
  return (ushort_t)(r >> 16);
}

static __device__ __forceinline__ v8f wmma_bf16(const Frag& a, const Frag& b, v8f c) {
  // (neg_a, A, neg_b, B, c_mod, C, reuse_a, reuse_b)
  return __builtin_amdgcn_wmma_f32_16x16x32_bf16(false, a.v, false, b.v,
                                                 (short)0, c, false, false);
}

// -------------------------- fp32 -> bf16 convert ---------------------------
__global__ __launch_bounds__(256) void cvt_bf16_kernel(
    const float* __restrict__ in, ushort_t* __restrict__ out, long long n) {
  long long i = (long long)blockIdx.x * blockDim.x + threadIdx.x;
  if (i < n) out[i] = f2bf(in[i]);
}

// ---------------------- GEMM: C[M,N] = A[M,K] * W[N,K]^T -------------------
// block = 256 thr (8 waves, 2x4), block tile 128(M) x 256(N), wave tile 64x64
// (4x4 WMMA frags), K-step 32.  Fragments loaded straight from global
// (L2-resident) in the ISA-mandated WMMA layouts:
//   A 16x32 bf16: lane row = lane%16, halves at K {half*8..+8, 16+half*8..+8}
//   B 32x16 bf16: lane col = lane%16 (row n of W), halves at K {half*8..+8, 16+half*8..+8}
__global__ __launch_bounds__(256) void gemm_bf16_kernel(
    const ushort_t* __restrict__ A, const ushort_t* __restrict__ W,
    float* __restrict__ C, int M, int N, int K) {
  const int lane = threadIdx.x & 31;
  const int wid  = threadIdx.x >> 5;
  const int half = lane >> 4;          // lane group 0/1
  const int l16  = lane & 15;
  const int mblk = blockIdx.y * 128 + (wid >> 2) * 64;
  const int nblk = blockIdx.x * 256 + (wid & 3) * 64;

  v8f c[4][4];
#pragma unroll
  for (int i = 0; i < 4; ++i)
#pragma unroll
    for (int j = 0; j < 4; ++j) c[i][j] = {};

  const ushort_t* arow = A + (size_t)(mblk + l16) * K + half * 8;
  const ushort_t* brow = W + (size_t)(nblk + l16) * K + half * 8;

  for (int k0 = 0; k0 < K; k0 += 32) {
    Frag a[4], b[4];
#pragma unroll
    for (int i = 0; i < 4; ++i) {
      const ushort_t* ar = arow + (size_t)(i * 16) * K + k0;
      a[i].q[0] = *(const u32x4*)(ar);
      a[i].q[1] = *(const u32x4*)(ar + 16);
    }
#pragma unroll
    for (int j = 0; j < 4; ++j) {
      const ushort_t* br = brow + (size_t)(j * 16) * K + k0;
      b[j].q[0] = *(const u32x4*)(br);
      b[j].q[1] = *(const u32x4*)(br + 16);
    }
#pragma unroll
    for (int i = 0; i < 4; ++i)
#pragma unroll
      for (int j = 0; j < 4; ++j)
        c[i][j] = wmma_bf16(a[i], b[j], c[i][j]);
  }

  // C f32 layout: row m = r + 8*half, col n = l16
#pragma unroll
  for (int i = 0; i < 4; ++i)
#pragma unroll
    for (int r = 0; r < 8; ++r) {
      const size_t m0 = (size_t)(mblk + i * 16 + half * 8 + r);
      float* crow = C + m0 * N + nblk + l16;
#pragma unroll
      for (int j = 0; j < 4; ++j) crow[j * 16] = c[i][j][r];
    }
}

// -------------------- RoPE + L2 norm + layout for attention ----------------
// one wave per (b,s,h).  q -> qb[B,H,S,HD]*attn_scale, k -> kb[B,H,S,HD],
// v -> vT[B,H,HD,S]  (transposed so p@v B-fragments are contiguous B128).
__global__ __launch_bounds__(256) void rope_norm_kernel(
    const float* __restrict__ qkv, const float* __restrict__ fc,
    const float* __restrict__ attn_scale_p,
    ushort_t* __restrict__ qb, ushort_t* __restrict__ kb,
    ushort_t* __restrict__ vT) {
  const int lane = threadIdx.x & 31;
  const int wid  = threadIdx.x >> 5;
  const int g = blockIdx.x * 8 + wid;            // (b,s,h) flat, h fastest
  const int h = g % Hc;
  const int s = (g / Hc) % Sc;
  const int b = g / (Hc * Sc);

  const float* qrow = qkv + ((size_t)(b * Sc + s)) * (3 * Dc) + h * HDc;
  const float* krow = qrow + Dc;
  const float* vrow = qrow + 2 * Dc;
  const float* f    = fc + (size_t)s * HDc;      // [HD/2][2] cos,sin
  const float scale = attn_scale_p[0];

  float qre[2], qim[2], kre[2], kim[2];
  float qss = 0.f, kss = 0.f;
#pragma unroll
  for (int t = 0; t < 2; ++t) {
    const int i = lane + t * 32;                 // pair index 0..63
    const float c = f[2 * i], sn = f[2 * i + 1];
    const float q0 = qrow[2 * i], q1 = qrow[2 * i + 1];
    const float k0 = krow[2 * i], k1 = krow[2 * i + 1];
    qre[t] = q0 * c - q1 * sn;  qim[t] = q0 * sn + q1 * c;
    kre[t] = k0 * c - k1 * sn;  kim[t] = k0 * sn + k1 * c;
    qss += qre[t] * qre[t] + qim[t] * qim[t];
    kss += kre[t] * kre[t] + kim[t] * kim[t];
  }
#pragma unroll
  for (int off = 16; off > 0; off >>= 1) {
    qss += __shfl_xor(qss, off, 32);
    kss += __shfl_xor(kss, off, 32);
  }
  const float qinv = scale / (sqrtf(qss) + EPSc);
  const float kinv = 1.0f  / (sqrtf(kss) + EPSc);

  ushort_t* qd = qb + (((size_t)(b * Hc + h)) * Sc + s) * HDc;
  ushort_t* kd = kb + (((size_t)(b * Hc + h)) * Sc + s) * HDc;
#pragma unroll
  for (int t = 0; t < 2; ++t) {
    const int i = lane + t * 32;
    qd[2 * i]     = f2bf(qre[t] * qinv);
    qd[2 * i + 1] = f2bf(qim[t] * qinv);
    kd[2 * i]     = f2bf(kre[t] * kinv);
    kd[2 * i + 1] = f2bf(kim[t] * kinv);
  }
  ushort_t* vd = vT + ((size_t)(b * Hc + h)) * HDc * Sc;   // [HD][S]
#pragma unroll
  for (int t = 0; t < 4; ++t) {
    const int d = lane * 4 + t;
    vd[(size_t)d * Sc + s] = f2bf(vrow[d]);
  }
}

// ----------------------------- flash attention -----------------------------
// one wave per 16-row q tile; key blocks of 32; online softmax in fp32.
// scores: 8 WMMA (2 key sub-tiles x 4 d-chunks); p@v: 8 WMMA (8 d tiles).
// P transposed C-layout -> A-layout via 1KB/wave LDS (bf16).
// q-tiles interleaved across blocks (stride 16) to balance the causal
// triangle's work across WGPs.
__global__ __launch_bounds__(256) void flash_attn_kernel(
    const ushort_t* __restrict__ qb, const ushort_t* __restrict__ kb,
    const ushort_t* __restrict__ vT, ushort_t* __restrict__ attnb) {
  __shared__ alignas(16) ushort_t pbuf[8][16 * 32];

  const int lane = threadIdx.x & 31;
  const int wid  = threadIdx.x >> 5;
  const int half = lane >> 4;
  const int l16  = lane & 15;

  const int bh  = blockIdx.x >> 4;               // b*H + h   (grid = 32*16)
  const int grp = blockIdx.x & 15;
  const int b   = bh >> 4;
  const int h   = bh & (Hc - 1);
  const int qbase = (grp + wid * 16) * 16;       // interleaved load balance

  const ushort_t* qptr = qb + ((size_t)bh * Sc + qbase) * HDc;
  const ushort_t* kptr = kb + (size_t)bh * Sc * HDc;
  const ushort_t* vptr = vT + (size_t)bh * HDc * Sc;

  // Q A-fragments over d: 4 x (16x32)
  Frag aq[4];
#pragma unroll
  for (int t = 0; t < 4; ++t) {
    const ushort_t* p0 = qptr + (size_t)l16 * HDc + t * 32 + half * 8;
    aq[t].q[0] = *(const u32x4*)(p0);
    aq[t].q[1] = *(const u32x4*)(p0 + 16);
  }

  v8f acc[8];
#pragma unroll
  for (int t = 0; t < 8; ++t) acc[t] = {};
  float mrun[8], lrun[8];
#pragma unroll
  for (int r = 0; r < 8; ++r) { mrun[r] = -1e30f; lrun[r] = 0.f; }

  for (int kb0 = 0; kb0 <= qbase + 15; kb0 += 32) {
    // ---- scores: S[16q x 32k] = Q (16x128) . K^T ----
    v8f s0 = {}, s1 = {};
#pragma unroll
    for (int t = 0; t < 4; ++t) {
      Frag bk0, bk1;
      const ushort_t* kp = kptr + (size_t)(kb0 + l16) * HDc + t * 32 + half * 8;
      bk0.q[0] = *(const u32x4*)(kp);
      bk0.q[1] = *(const u32x4*)(kp + 16);
      const ushort_t* kp2 = kp + (size_t)16 * HDc;
      bk1.q[0] = *(const u32x4*)(kp2);
      bk1.q[1] = *(const u32x4*)(kp2 + 16);
      s0 = wmma_bf16(aq[t], bk0, s0);
      s1 = wmma_bf16(aq[t], bk1, s1);
    }

    // ---- causal mask + online softmax stats (fp32) ----
    const int key0 = kb0 + l16, key1 = key0 + 16;
    float p0[8], p1[8], sc[8];
#pragma unroll
    for (int r = 0; r < 8; ++r) {
      const int qi = qbase + half * 8 + r;
      float v0 = (key0 <= qi) ? s0[r] : -1e30f;
      float v1 = (key1 <= qi) ? s1[r] : -1e30f;
      float rm = fmaxf(v0, v1);
#pragma unroll
      for (int off = 8; off > 0; off >>= 1)      // reduce within 16-lane half
        rm = fmaxf(rm, __shfl_xor(rm, off, 32));
      const float nm = fmaxf(mrun[r], rm);
      sc[r] = __expf(mrun[r] - nm);
      mrun[r] = nm;
      p0[r] = __expf(v0 - nm);
      p1[r] = __expf(v1 - nm);
      float rs = p0[r] + p1[r];
#pragma unroll
      for (int off = 8; off > 0; off >>= 1)
        rs += __shfl_xor(rs, off, 32);
      lrun[r] = lrun[r] * sc[r] + rs;
    }

    // ---- P: C-layout -> A-layout through per-wave LDS (bf16) ----
    ushort_t* pl = pbuf[wid];
#pragma unroll
    for (int r = 0; r < 8; ++r) {
      const int m = half * 8 + r;
      pl[m * 32 + l16]      = f2bf(p0[r]);
      pl[m * 32 + 16 + l16] = f2bf(p1[r]);
    }
    asm volatile("s_wait_dscnt 0" ::: "memory");   // per-wave LDS RAW fence
    Frag ap;
    ap.q[0] = *(const u32x4*)(pl + l16 * 32 + half * 8);
    ap.q[1] = *(const u32x4*)(pl + l16 * 32 + 16 + half * 8);

    // ---- rescale accumulators, then out += P (16x32) . V (32x16 per d tile)
#pragma unroll
    for (int t = 0; t < 8; ++t) {
#pragma unroll
      for (int r = 0; r < 8; ++r) acc[t][r] *= sc[r];
      Frag bv;
      const ushort_t* vp = vptr + (size_t)(t * 16 + l16) * Sc + kb0 + half * 8;
      bv.q[0] = *(const u32x4*)(vp);
      bv.q[1] = *(const u32x4*)(vp + 16);
      acc[t] = wmma_bf16(ap, bv, acc[t]);
    }
  }

  // ---- epilogue: normalize by row sum, store bf16 to attnb[B,S,D] ----
  ushort_t* orow = attnb + ((size_t)b * Sc + qbase) * Dc + h * HDc;
#pragma unroll
  for (int r = 0; r < 8; ++r) {
    const float linv = 1.0f / lrun[r];
    const int m = half * 8 + r;
#pragma unroll
    for (int t = 0; t < 8; ++t)
      orow[(size_t)m * Dc + t * 16 + l16] = f2bf(acc[t][r] * linv);
  }
}

// ------------------------------- launcher ----------------------------------
extern "C" void kernel_launch(void* const* d_in, const int* in_sizes, int n_in,
                              void* d_out, int out_size, void* d_ws, size_t ws_size,
                              hipStream_t stream) {
  const float* x    = (const float*)d_in[0];   // [B,S,D]
  const float* fc   = (const float*)d_in[1];   // [S,HD/2,2]
  const float* wqkv = (const float*)d_in[2];   // [3D,D]
  const float* wout = (const float*)d_in[3];   // [D,D]
  const float* ascl = (const float*)d_in[4];   // [1]
  float* out = (float*)d_out;                  // [B,S,D] fp32

  char* ws = (char*)d_ws;
  const long long nX  = (long long)Bc * Sc * Dc;       // 8,388,608
  const long long nWq = (long long)3 * Dc * Dc;        // 12,582,912
  const long long nWo = (long long)Dc * Dc;            // 4,194,304
  const long long nBHSD = (long long)Bc * Hc * Sc * HDc;

  size_t off = 0;
  ushort_t* xb   = (ushort_t*)(ws + off); off += (size_t)nX  * 2;
  ushort_t* wqb  = (ushort_t*)(ws + off); off += (size_t)nWq * 2;
  ushort_t* wob  = (ushort_t*)(ws + off); off += (size_t)nWo * 2;
  float*    qkvf = (float*)   (ws + off); off += (size_t)Bc * Sc * 3 * Dc * 4;
  ushort_t* qbuf = (ushort_t*)(ws + off); off += (size_t)nBHSD * 2;
  ushort_t* kbuf = (ushort_t*)(ws + off); off += (size_t)nBHSD * 2;
  ushort_t* vTb  = (ushort_t*)(ws + off); off += (size_t)nBHSD * 2;
  ushort_t* attb = (ushort_t*)(ws + off); off += (size_t)nX * 2;
  (void)ws_size; // ~218 MB needed

  // 1) fp32 -> bf16 conversions
  cvt_bf16_kernel<<<(unsigned)((nX  + 255) / 256), 256, 0, stream>>>(x,    xb,  nX);
  cvt_bf16_kernel<<<(unsigned)((nWq + 255) / 256), 256, 0, stream>>>(wqkv, wqb, nWq);
  cvt_bf16_kernel<<<(unsigned)((nWo + 255) / 256), 256, 0, stream>>>(wout, wob, nWo);

  // 2) QKV GEMM: [4096 x 2048] * [6144 x 2048]^T -> fp32
  {
    dim3 grid(3 * Dc / 256, Bc * Sc / 128);
    gemm_bf16_kernel<<<grid, 256, 0, stream>>>(xb, wqb, qkvf,
                                               Bc * Sc, 3 * Dc, Dc);
  }
  // 3) RoPE + L2 norm + attn_scale + v transpose
  rope_norm_kernel<<<Bc * Sc * Hc / 8, 256, 0, stream>>>(qkvf, fc, ascl,
                                                         qbuf, kbuf, vTb);
  // 4) causal flash attention (wave-per-qtile, 8 waves/block)
  flash_attn_kernel<<<Bc * Hc * 16, 256, 0, stream>>>(qbuf, kbuf, vTb, attb);

  // 5) output projection: [4096 x 2048] * [2048 x 2048]^T -> fp32 d_out
  {
    dim3 grid(Dc / 256, Bc * Sc / 128);
    gemm_bf16_kernel<<<grid, 256, 0, stream>>>(attb, wob, out,
                                               Bc * Sc, Dc, Dc);
  }
}
